// PerImageAttentionLayer_76759655514670
// MI455X (gfx1250) — compile-verified, matrix operand order checked by reading, not run on hardware
//
#include <hip/hip_runtime.h>
#include <hip/hip_bf16.h>

// ---------------------------------------------------------------------------
// Types for WMMA (CDNA5 / gfx1250, wave32)
// ---------------------------------------------------------------------------
typedef __bf16 bf16_t;
typedef __attribute__((ext_vector_type(8)))  __bf16 v8bf;
typedef __attribute__((ext_vector_type(16))) __bf16 v16bf;
typedef __attribute__((ext_vector_type(8)))  float  v8f;
typedef __attribute__((ext_vector_type(4)))  float  v4f;

#define B_ 2
#define N_ 8192
#define D_ 256
#define C_ 6
#define HF 64
#define WF 64
#define P_ 4
#define M1 (B_ * N_)   // 16384 query rows

__device__ __forceinline__ float wave_sum(float v) {
    #pragma unroll
    for (int o = 16; o > 0; o >>= 1) v += __shfl_xor(v, o, 32);
    return v;
}

// ---------------------------------------------------------------------------
// Weight folding (tiny): WqkT[n][i] = sum_o Wq[o][i]*Wk[o][n]  (bf16)
//                        Wvo [n][k] = sum_j Wo[n][j]*Wv[j][k]  (bf16)
// ---------------------------------------------------------------------------
__global__ void prep_mats(const float* __restrict__ Wq, const float* __restrict__ Wk,
                          const float* __restrict__ Wv, const float* __restrict__ Wo,
                          bf16_t* __restrict__ WqkT, bf16_t* __restrict__ Wvo) {
    const int n = blockIdx.x;     // 0..255
    const int i = threadIdx.x;    // 0..255
    float s0 = 0.f, s1 = 0.f;
    #pragma unroll 8
    for (int o = 0; o < D_; ++o) {
        s0 += Wq[o * D_ + i] * Wk[o * D_ + n];   // (Wq^T Wk)[i][n]
        s1 += Wo[n * D_ + o] * Wv[o * D_ + i];   // (Wo Wv)[n][i]
    }
    WqkT[n * D_ + i] = (bf16_t)s0;
    Wvo [n * D_ + i] = (bf16_t)s1;
}

// gbias[n]=bq@Wk[:,n] ; wvb[n]=Wo[n,:]@bv ; hqk[i]=Wq[:,i]@bk ; c0=bq.bk
__global__ void prep_vecs(const float* __restrict__ Wq, const float* __restrict__ bq,
                          const float* __restrict__ Wk, const float* __restrict__ bk,
                          const float* __restrict__ bv, const float* __restrict__ Wo,
                          float* __restrict__ gbias, float* __restrict__ wvb,
                          float* __restrict__ hqk, float* __restrict__ c0) {
    const int t = threadIdx.x;    // 0..255
    float sg = 0.f, sw = 0.f, sh = 0.f;
    #pragma unroll 8
    for (int o = 0; o < D_; ++o) {
        sg += bq[o] * Wk[o * D_ + t];
        sw += Wo[t * D_ + o] * bv[o];
        sh += Wq[o * D_ + t] * bk[o];
    }
    gbias[t] = sg; wvb[t] = sw; hqk[t] = sh;
    if (t == 0) {
        float s = 0.f;
        for (int o = 0; o < D_; ++o) s += bq[o] * bk[o];
        *c0 = s;
    }
}

// ---------------------------------------------------------------------------
// LayerNorm + offset head + attn_extra.  One wave32 per row; lane owns 8 ch.
// ---------------------------------------------------------------------------
__global__ void ln_offsets(const float* __restrict__ queries,
                           const float* __restrict__ ln_g, const float* __restrict__ ln_b,
                           const float* __restrict__ Woff, const float* __restrict__ boff,
                           const float* __restrict__ hqk, const float* __restrict__ c0,
                           bf16_t* __restrict__ qn_bf, float* __restrict__ offs,
                           float* __restrict__ extra) {
    const int wv   = threadIdx.x >> 5;
    const int lane = threadIdx.x & 31;
    const int row  = blockIdx.x * 8 + wv;
    const int ch   = lane * 8;
    const float* xr = queries + (size_t)row * D_ + ch;

    float x[8];
    *(v4f*)(x)     = *(const v4f*)(xr);
    *(v4f*)(x + 4) = *(const v4f*)(xr + 4);

    float s = x[0]+x[1]+x[2]+x[3]+x[4]+x[5]+x[6]+x[7];
    const float mu = wave_sum(s) * (1.f / D_);
    float s2 = 0.f;
    #pragma unroll
    for (int i = 0; i < 8; ++i) { float d = x[i] - mu; s2 += d * d; }
    const float inv = rsqrtf(wave_sum(s2) * (1.f / D_) + 1e-5f);

    float qn[8];
    #pragma unroll
    for (int i = 0; i < 8; ++i)
        qn[i] = (x[i] - mu) * inv * ln_g[ch + i] + ln_b[ch + i];

    bf16_t* qo = qn_bf + (size_t)row * D_ + ch;
    #pragma unroll
    for (int i = 0; i < 8; ++i) qo[i] = (bf16_t)qn[i];

    // 8 offset outputs: (qn . Woff[j]) + boff[j], scaled by 0.05
    #pragma unroll
    for (int j = 0; j < 2 * P_; ++j) {
        float p = 0.f;
        #pragma unroll
        for (int i = 0; i < 8; ++i) p += qn[i] * Woff[j * D_ + ch + i];
        p = wave_sum(p);
        if (lane == 0) offs[(size_t)row * (2 * P_) + j] = (p + boff[j]) * 0.05f;
    }
    // attn_extra = qn . hqk + bq.bk
    float pe = 0.f;
    #pragma unroll
    for (int i = 0; i < 8; ++i) pe += qn[i] * hqk[ch + i];
    pe = wave_sum(pe);
    if (lane == 0) extra[row] = pe + *c0;
}

// ---------------------------------------------------------------------------
// WMMA bf16 GEMM:  Y[M x 256] = X[M x 256] @ Wt^T  (+ bias / fused epilogue)
// Wt stored as 256 rows (output col) x 256 K, K contiguous.
// Block = 128 thr = 4 waves; wave computes a 32x64 strip:
//   2 A-row tiles x 4 B-col tiles = 8 accumulators.
// Per K-step: 4 A-loads + 8 B-loads feed 8 WMMAs (1.5 B128 loads / WMMA),
// halving L2 weight traffic vs a 16-row strip.
// ---------------------------------------------------------------------------
template<bool FUSED>
__global__ void gemm256_wmma(const bf16_t* __restrict__ X, const bf16_t* __restrict__ Wt,
                             const float* __restrict__ bias, float* __restrict__ Y,
                             const float* __restrict__ residual,
                             const float* __restrict__ srow,
                             const float* __restrict__ wvb) {
    const int m0   = blockIdx.x * 32;
    const int wv   = threadIdx.x >> 5;
    const int lane = threadIdx.x & 31;
    const int n0   = wv * 64;
    const int rA   = lane & 15;     // A row / B column within tile
    const int hi   = lane >> 4;     // K half select

    v8f acc[2][4];
    #pragma unroll
    for (int t = 0; t < 2; ++t)
        #pragma unroll
        for (int j = 0; j < 4; ++j) acc[t][j] = (v8f){};

    const bf16_t* xrow0 = X + (size_t)(m0 + rA) * D_ + hi * 8;
    const bf16_t* xrow1 = xrow0 + 16 * D_;

    #pragma unroll
    for (int k = 0; k < D_; k += 32) {
        v8bf a0lo = *(const v8bf*)(xrow0 + k);
        v8bf a0hi = *(const v8bf*)(xrow0 + k + 16);
        v8bf a1lo = *(const v8bf*)(xrow1 + k);
        v8bf a1hi = *(const v8bf*)(xrow1 + k + 16);
        v16bf a0 = __builtin_shufflevector(a0lo, a0hi, 0,1,2,3,4,5,6,7,8,9,10,11,12,13,14,15);
        v16bf a1 = __builtin_shufflevector(a1lo, a1hi, 0,1,2,3,4,5,6,7,8,9,10,11,12,13,14,15);
        #pragma unroll
        for (int j = 0; j < 4; ++j) {
            const bf16_t* wrow = Wt + (size_t)(n0 + j * 16 + rA) * D_ + hi * 8 + k;
            v8bf blo = *(const v8bf*)(wrow);
            v8bf bhi = *(const v8bf*)(wrow + 16);
            v16bf bm = __builtin_shufflevector(blo, bhi, 0,1,2,3,4,5,6,7,8,9,10,11,12,13,14,15);
            acc[0][j] = __builtin_amdgcn_wmma_f32_16x16x32_bf16(
                false, a0, false, bm, (short)0, acc[0][j], false, false);
            acc[1][j] = __builtin_amdgcn_wmma_f32_16x16x32_bf16(
                false, a1, false, bm, (short)0, acc[1][j], false, false);
        }
    }

    // C/D layout: lanes 0-15 -> M = r, lanes 16-31 -> M = 8 + r ; N = lane&15
    #pragma unroll
    for (int t = 0; t < 2; ++t) {
        #pragma unroll
        for (int j = 0; j < 4; ++j) {
            #pragma unroll
            for (int r = 0; r < 8; ++r) {
                const int m   = m0 + t * 16 + hi * 8 + r;
                const int col = n0 + j * 16 + rA;
                const size_t idx = (size_t)m * D_ + col;
                float y = acc[t][j][r] + bias[col];
                if (FUSED) y += residual[idx] + srow[m] * wvb[col];
                Y[idx] = y;
            }
        }
    }
}

// ---------------------------------------------------------------------------
// Fused bilinear sampling + per-camera attention accumulation.
// One wave32 per (b,n); never materializes sampled / k / v.
//   u[b,n,:] = (1/W) * sum_c attn_c * sm_c ;  srow = (1/W) * sum_c attn_c
// ---------------------------------------------------------------------------
__global__ void sample_attn(const float* __restrict__ feats,      // (B,C,64,64,256)
                            const float* __restrict__ pix,        // (B,C,N,2)
                            const int*   __restrict__ vmask,      // (B,C,N)
                            const float* __restrict__ offs,       // (M1,P,2)
                            const float* __restrict__ g,          // (M1,256)
                            const float* __restrict__ extra,      // (M1)
                            bf16_t* __restrict__ u_bf,            // (M1,256)
                            float* __restrict__ srow) {           // (M1)
    const int wv   = threadIdx.x >> 5;
    const int lane = threadIdx.x & 31;
    const int row  = blockIdx.x * 8 + wv;     // (b,n) flat
    const int b    = row / N_;
    const int n    = row - b * N_;
    const int ch   = lane * 8;

    float gv[8];
    {
        const float* gr = g + (size_t)row * D_ + ch;
        *(v4f*)(gv)     = *(const v4f*)(gr);
        *(v4f*)(gv + 4) = *(const v4f*)(gr + 4);
    }
    const float ex = extra[row];

    float ox[P_], oy[P_];
    #pragma unroll
    for (int p = 0; p < P_; ++p) {
        ox[p] = offs[((size_t)row * P_ + p) * 2 + 0];
        oy[p] = offs[((size_t)row * P_ + p) * 2 + 1];
    }

    float u[8] = {0, 0, 0, 0, 0, 0, 0, 0};
    float s_a = 0.f;
    int wsum = 0;

    for (int c = 0; c < C_; ++c) {
        const size_t cn = ((size_t)b * C_ + c) * N_ + n;
        const int valid = vmask[cn];
        wsum += (valid != 0);
        if (!valid) continue;

        const float px = pix[cn * 2 + 0];
        const float py = pix[cn * 2 + 1];
        const float* feat = feats + ((size_t)b * C_ + c) * (HF * WF * D_);

        float sm[8] = {0, 0, 0, 0, 0, 0, 0, 0};
        #pragma unroll
        for (int p = 0; p < P_; ++p) {
            const float xf = (px + ox[p] + 1.f) * 0.5f * (WF - 1);
            const float yf = (py + oy[p] + 1.f) * 0.5f * (HF - 1);
            const float x0f = floorf(xf), y0f = floorf(yf);
            const float wx1 = xf - x0f, wx0 = 1.f - wx1;
            const float wy1 = yf - y0f, wy0 = 1.f - wy1;
            const int x0 = (int)x0f, y0 = (int)y0f;

            #pragma unroll
            for (int cor = 0; cor < 4; ++cor) {
                const int xi = x0 + (cor & 1);
                const int yi = y0 + (cor >> 1);
                const float w = ((cor & 1) ? wx1 : wx0) * ((cor >> 1) ? wy1 : wy0);
                if (xi >= 0 && xi < WF && yi >= 0 && yi < HF) {
                    const float* fr = feat + ((size_t)yi * WF + xi) * D_ + ch;
                    float f[8];
                    *(v4f*)(f)     = *(const v4f*)(fr);
                    *(v4f*)(f + 4) = *(const v4f*)(fr + 4);
                    #pragma unroll
                    for (int i = 0; i < 8; ++i) sm[i] = fmaf(w, f[i], sm[i]);
                }
            }
        }
        #pragma unroll
        for (int i = 0; i < 8; ++i) sm[i] *= (1.f / P_);

        float pd = 0.f;
        #pragma unroll
        for (int i = 0; i < 8; ++i) pd += gv[i] * sm[i];
        const float attn = (wave_sum(pd) + ex) * 0.0625f;   // /sqrt(256)

        #pragma unroll
        for (int i = 0; i < 8; ++i) u[i] = fmaf(attn, sm[i], u[i]);
        s_a += attn;
    }

    const float invW = 1.f / fmaxf((float)wsum, 1.f);
    bf16_t* uo = u_bf + (size_t)row * D_ + ch;
    #pragma unroll
    for (int i = 0; i < 8; ++i) uo[i] = (bf16_t)(u[i] * invW);
    if (lane == 0) srow[row] = s_a * invW;
}

// ---------------------------------------------------------------------------
// Launch
// ---------------------------------------------------------------------------
extern "C" void kernel_launch(void* const* d_in, const int* in_sizes, int n_in,
                              void* d_out, int out_size, void* d_ws, size_t ws_size,
                              hipStream_t stream) {
    const float* queries = (const float*)d_in[0];
    const float* feats   = (const float*)d_in[1];
    const float* pix     = (const float*)d_in[2];
    const int*   vmask   = (const int*)  d_in[3];
    const float* Wq = (const float*)d_in[4];  const float* bq = (const float*)d_in[5];
    const float* Wk = (const float*)d_in[6];  const float* bk = (const float*)d_in[7];
    const float* Wv = (const float*)d_in[8];  const float* bv = (const float*)d_in[9];
    const float* Wo = (const float*)d_in[10]; const float* bo = (const float*)d_in[11];
    const float* Woff = (const float*)d_in[12]; const float* boff = (const float*)d_in[13];
    const float* ln_g = (const float*)d_in[14]; const float* ln_b = (const float*)d_in[15];
    float* out = (float*)d_out;

    // ---- workspace carve ----
    char* w = (char*)d_ws;
    auto carve = [&](size_t bytes) { char* p = w; w += (bytes + 255) & ~(size_t)255; return p; };
    bf16_t* qn_bf = (bf16_t*)carve((size_t)M1 * D_ * sizeof(bf16_t));   // 8.4 MB
    bf16_t* u_bf  = (bf16_t*)carve((size_t)M1 * D_ * sizeof(bf16_t));   // 8.4 MB
    float*  gmat  = (float*) carve((size_t)M1 * D_ * sizeof(float));    // 16.8 MB
    float*  offs  = (float*) carve((size_t)M1 * 2 * P_ * sizeof(float));
    float*  extra = (float*) carve((size_t)M1 * sizeof(float));
    float*  srow  = (float*) carve((size_t)M1 * sizeof(float));
    bf16_t* WqkT  = (bf16_t*)carve((size_t)D_ * D_ * sizeof(bf16_t));
    bf16_t* Wvo   = (bf16_t*)carve((size_t)D_ * D_ * sizeof(bf16_t));
    float*  gbias = (float*) carve(D_ * sizeof(float));
    float*  wvb   = (float*) carve(D_ * sizeof(float));
    float*  hqk   = (float*) carve(D_ * sizeof(float));
    float*  c0    = (float*) carve(sizeof(float));

    // 1. fold weights
    prep_mats<<<D_, D_, 0, stream>>>(Wq, Wk, Wv, Wo, WqkT, Wvo);
    prep_vecs<<<1, D_, 0, stream>>>(Wq, bq, Wk, bk, bv, Wo, gbias, wvb, hqk, c0);

    // 2. layernorm + offsets + attn_extra
    ln_offsets<<<M1 / 8, 256, 0, stream>>>(queries, ln_g, ln_b, Woff, boff,
                                           hqk, c0, qn_bf, offs, extra);

    // 3. g = qn @ (Wq^T Wk) + gbias      (WMMA bf16, 32-row wave strips)
    gemm256_wmma<false><<<M1 / 32, 128, 0, stream>>>(qn_bf, WqkT, gbias, gmat,
                                                     nullptr, nullptr, nullptr);

    // 4. fused deformable sampling + attention accumulation
    sample_attn<<<M1 / 8, 256, 0, stream>>>(feats, pix, vmask, offs, gmat, extra,
                                            u_bf, srow);

    // 5. out = residual + u @ (Wo Wv)^T + srow*wvb + bo   (WMMA bf16, fused epilogue)
    gemm256_wmma<true><<<M1 / 32, 128, 0, stream>>>(u_bf, Wvo, bo, out,
                                                    queries, srow, wvb);
}
